// CrossAttention_60816736911806
// MI455X (gfx1250) — matrix operand order repaired and checked by reading, not compiled, not run on hardware
//
#include <hip/hip_runtime.h>

// ---------------------------------------------------------------------------
// CDNA5 (gfx1250) cross-attention, wave32 + WMMA.
//   - V_WMMA_F32_16X16X4_F32   for attention (hd==4, fp32 softmax path)
//   - V_WMMA_F32_16X16X32_BF16 for all pointwise convs (fp32 accumulate)
// ---------------------------------------------------------------------------

typedef __attribute__((ext_vector_type(2)))  float  v2f;
typedef __attribute__((ext_vector_type(8)))  float  v8f;
typedef __attribute__((ext_vector_type(16))) __bf16 v16bf;

__device__ __forceinline__ unsigned short f2bf(float f) {
  unsigned u = __float_as_uint(f);
  u += 0x7FFFu + ((u >> 16) & 1u);           // round-to-nearest-even
  return (unsigned short)(u >> 16);
}

// 16-bit A-matrix (16x32) element->K map: lanes 0-15 rows hold K 0..15;
// lanes 16-31 hold K 16..31 (matches the B-operand LDS layout).
__device__ __forceinline__ int amapK(int e, int lane) {
  return e + ((lane >> 4) & 1) * 16;
}

__device__ __forceinline__ v8f wmma_bf16(v16bf a, v16bf b, v8f c) {
  return __builtin_amdgcn_wmma_f32_16x16x32_bf16(false, a, false, b, (short)0, c, false, false);
}
__device__ __forceinline__ v8f wmma_f32k4(v2f a, v2f b, v8f c) {
  return __builtin_amdgcn_wmma_f32_16x16x4_f32(false, a, false, b, (short)0, c, false, false);
}

// ---------------------------------------------------------------------------
// Kernel 0: fold BN scale into conv weight, convert to bf16, and pre-swizzle
// into WMMA A-operand order: chunk(mt,kb) = 512 elems, elem = lane*16 + e,
// value = W[mt*16 + (lane&15)][kb*32 + amapK(e,lane)] * s[row].
// ---------------------------------------------------------------------------
__global__ void k_prep_w(const float* __restrict__ W, const float* __restrict__ s,
                         unsigned short* __restrict__ out, int O, int I) {
  int tid = blockIdx.x * blockDim.x + threadIdx.x;
  if (tid >= O * I) return;
  int chunk = tid >> 9;
  int r     = tid & 511;
  int lane  = r >> 4;
  int e     = r & 15;
  int nkb   = I >> 5;
  int mt    = chunk / nkb;
  int kb    = chunk - mt * nkb;
  int row   = mt * 16 + (lane & 15);
  int col   = kb * 32 + amapK(e, lane);
  out[tid]  = f2bf(W[row * I + col] * s[row]);
}

// ---------------------------------------------------------------------------
// Projection: _t = bn(Wc @ in)  (M=32, K=C, N=128/block), then Q/K/V = bn(Wp @ _t)
// Input tile lives in LDS in bf16 B-operand layout (lanes 0-15: K 0..15,
// lanes 16-31: K 16..31, N = lane&15). 8 waves, wave = n-subtile.
// ---------------------------------------------------------------------------
__global__ void __launch_bounds__(256)
k_proj(const float* __restrict__ xin, const unsigned short* __restrict__ wsw,
       int wc_base, int wq_base, int wk_base, int wv_base,
       const float* __restrict__ pbc, const float* __restrict__ pbq,
       const float* __restrict__ pbk, const float* __restrict__ pbv,
       float* __restrict__ Qo, float* __restrict__ Ko, float* __restrict__ Vto,
       int C, int Npix) {
  extern __shared__ unsigned short smem_u16[];
  const int nkb = C >> 5;
  unsigned short* sX = smem_u16;                    // C x 128 tile, swizzled
  unsigned short* sT = smem_u16 + nkb * 8 * 512;    // 32 x 128 _t tile, swizzled

  const int tid  = threadIdx.x;
  const int lane = tid & 31;
  const int wid  = tid >> 5;
  const int half = lane >> 4;
  const int nl   = lane & 15;
  const int p0   = blockIdx.x * 128;
  const int b    = blockIdx.y;

  // cooperative load + bf16 swizzle of the input tile
  const float* xb = xin + (size_t)b * C * Npix;
  for (int i = tid; i < C * 128; i += 256) {
    int pl  = i & 127;
    int c   = i >> 7;
    float val = xb[(size_t)c * Npix + p0 + pl];
    int kb = c >> 5, kk = c & 31;
    int nt = pl >> 4, pnl = pl & 15;
    sX[(kb * 8 + nt) * 512 + (pnl + ((kk >= 16) ? 16 : 0)) * 16 + (kk & 15)] = f2bf(val);
  }
  __syncthreads();

  const int nt = wid;
  // stage 1: 32 x 16 tile of _t, K-loop over bf16 WMMAs
  v8f acc0 = {}; v8f acc1 = {};
  for (int kb = 0; kb < nkb; ++kb) {
    v16bf bm = *(const v16bf*)(sX + (kb * 8 + nt) * 512 + lane * 16);
    v16bf a0 = *(const v16bf*)(wsw + wc_base + (0 * nkb + kb) * 512 + lane * 16);
    v16bf a1 = *(const v16bf*)(wsw + wc_base + (1 * nkb + kb) * 512 + lane * 16);
    acc0 = wmma_bf16(a0, bm, acc0);
    acc1 = wmma_bf16(a1, bm, acc1);
  }
  // epilogue: +bias, bf16, into sT in B-operand layout (single K=32 chunk / nt)
  #pragma unroll
  for (int v = 0; v < 8; ++v) {
    int r0 = v + half * 8;            // D layout: row = vgpr + 8*lanehi
    int r1 = 16 + r0;
    sT[nt * 512 + (nl +  0) * 16 + (r0 & 15)] = f2bf(acc0[v] + pbc[r0]);
    sT[nt * 512 + (nl + 16) * 16 + (r1 & 15)] = f2bf(acc1[v] + pbc[r1]);
  }
  __syncthreads();

  v16bf bt = *(const v16bf*)(sT + nt * 512 + lane * 16);
  const int p = p0 + nt * 16 + nl;
  v8f z = {};

  // Q  -> (b, 32, N) row-major
  {
    v16bf a0 = *(const v16bf*)(wsw + wq_base + lane * 16);
    v16bf a1 = *(const v16bf*)(wsw + wq_base + 512 + lane * 16);
    v8f c0 = wmma_bf16(a0, bt, z);
    v8f c1 = wmma_bf16(a1, bt, z);
    #pragma unroll
    for (int v = 0; v < 8; ++v) {
      int r0 = v + half * 8, r1 = 16 + r0;
      Qo[((size_t)b * 32 + r0) * Npix + p] = c0[v] + pbq[r0];
      Qo[((size_t)b * 32 + r1) * Npix + p] = c1[v] + pbq[r1];
    }
  }
  // K  -> (b, 32, N) row-major
  {
    v16bf a0 = *(const v16bf*)(wsw + wk_base + lane * 16);
    v16bf a1 = *(const v16bf*)(wsw + wk_base + 512 + lane * 16);
    v8f c0 = wmma_bf16(a0, bt, z);
    v8f c1 = wmma_bf16(a1, bt, z);
    #pragma unroll
    for (int v = 0; v < 8; ++v) {
      int r0 = v + half * 8, r1 = 16 + r0;
      Ko[((size_t)b * 32 + r0) * Npix + p] = c0[v] + pbk[r0];
      Ko[((size_t)b * 32 + r1) * Npix + p] = c1[v] + pbk[r1];
    }
  }
  // V  -> transposed (b, head, N, hd) for coalesced PV B-operand builds
  {
    v16bf a0 = *(const v16bf*)(wsw + wv_base + lane * 16);
    v16bf a1 = *(const v16bf*)(wsw + wv_base + 512 + lane * 16);
    v8f c0 = wmma_bf16(a0, bt, z);
    v8f c1 = wmma_bf16(a1, bt, z);
    #pragma unroll
    for (int v = 0; v < 8; ++v) {
      int r0 = v + half * 8, r1 = 16 + r0;
      Vto[(((size_t)b * 8 + (r0 >> 2)) * Npix + p) * 4 + (r0 & 3)] = c0[v] + pbv[r0];
      Vto[(((size_t)b * 8 + (r1 >> 2)) * Npix + p) * 4 + (r1 & 3)] = c1[v] + pbv[r1];
    }
  }
}

// ---------------------------------------------------------------------------
// Flash cross-attention, hd=4. One 16-query tile per wave, online softmax
// over Nk in 16-key tiles. Logits: one f32 16x16x4 WMMA; PV: four K=4 WMMAs
// with P routed through a private 1KB LDS tile (D-layout -> A-layout).
// K/V stream from global (fully L2-resident: 192MB L2 >> 5MB of QKV).
// ---------------------------------------------------------------------------
__global__ void __launch_bounds__(256)
k_attn(const float* __restrict__ Q, const float* __restrict__ K,
       const float* __restrict__ Vt, float* __restrict__ att, int Nq, int Nk) {
  __shared__ float sP[8 * 256];
  const int tid  = threadIdx.x;
  const int lane = tid & 31;
  const int wid  = tid >> 5;
  const int half = lane >> 4;
  const int nl   = lane & 15;
  const int h    = blockIdx.y;
  const int b    = blockIdx.z;
  const int qbase = blockIdx.x * 128 + wid * 16;
  const float GAMMA = 0.5f;                        // hd^-0.5, hd=4

  const float* Qh = Q  + ((size_t)b * 32 + h * 4) * Nq;
  const float* Kh = K  + ((size_t)b * 32 + h * 4) * Nk;
  const float* Vb = Vt + (((size_t)b * 8 + h) * (size_t)Nk) * 4;
  float* Pw = sP + wid * 256;

  // A operand (16x4 f32): lane row = nl, K = 2*half + e; GAMMA folded in
  v2f aq;
  #pragma unroll
  for (int e = 0; e < 2; ++e)
    aq[e] = Qh[(size_t)(2 * half + e) * Nq + qbase + nl] * GAMMA;

  float ml[8], ll[8];
  #pragma unroll
  for (int v = 0; v < 8; ++v) { ml[v] = -3.0e38f; ll[v] = 0.f; }
  v8f o = {};
  v8f z = {};

  for (int kt = 0; kt < (Nk >> 4); ++kt) {
    const int kb0 = kt * 16;
    // B operand (4x16 f32): lane col = key nl, K = 2*half + e
    v2f bk;
    #pragma unroll
    for (int e = 0; e < 2; ++e)
      bk[e] = Kh[(size_t)(2 * half + e) * Nk + kb0 + nl];
    v8f s = wmma_f32k4(aq, bk, z);

    // row-wise max: reduce over the 16 key lanes of each half
    float mx[8];
    #pragma unroll
    for (int v = 0; v < 8; ++v) mx[v] = s[v];
    #pragma unroll
    for (int d = 1; d <= 8; d <<= 1)
      #pragma unroll
      for (int v = 0; v < 8; ++v)
        mx[v] = fmaxf(mx[v], __shfl_xor(mx[v], d, 32));

    float pn[8], rs[8];
    #pragma unroll
    for (int v = 0; v < 8; ++v) {
      float mn = fmaxf(ml[v], mx[v]);
      float sc = __expf(ml[v] - mn);
      pn[v] = __expf(s[v] - mn);
      ml[v] = mn;
      ll[v] *= sc;
      o[v]  *= sc;                  // same row layout as stats: row = v + 8*half
      rs[v] = pn[v];
    }
    #pragma unroll
    for (int d = 1; d <= 8; d <<= 1)
      #pragma unroll
      for (int v = 0; v < 8; ++v)
        rs[v] += __shfl_xor(rs[v], d, 32);
    #pragma unroll
    for (int v = 0; v < 8; ++v) {
      ll[v] += rs[v];
      Pw[(v + half * 8) * 16 + nl] = pn[v];   // P tile in (row,col) order
    }
    // PV: O(16 x hd) += P(16x16) @ Vtile(16x4), as 4 chained K=4 WMMAs
    #pragma unroll
    for (int k0 = 0; k0 < 16; k0 += 4) {
      v2f ap = *(const v2f*)(Pw + nl * 16 + k0 + half * 2);   // contiguous pair
      v2f bv;
      #pragma unroll
      for (int e = 0; e < 2; ++e) {
        int krel = k0 + 2 * half + e;
        bv[e] = (nl < 4) ? Vb[(size_t)(kb0 + krel) * 4 + nl] : 0.f;
      }
      o = wmma_f32k4(ap, bv, o);
    }
  }
  #pragma unroll
  for (int v = 0; v < 8; ++v) o[v] *= 1.0f / ll[v];
  if (nl < 4) {                                     // columns 0..3 = hd
    #pragma unroll
    for (int v = 0; v < 8; ++v)
      att[((size_t)b * 32 + h * 4 + nl) * Nq + qbase + v + half * 8] = o[v];
  }
}

// ---------------------------------------------------------------------------
// Output conv (Cout x 32 @ 32 x N) + bias + residual, bf16 WMMA, K=32.
// ---------------------------------------------------------------------------
__global__ void __launch_bounds__(256)
k_out(const float* __restrict__ att, const unsigned short* __restrict__ wsw, int wbase,
      const float* __restrict__ bias, const float* __restrict__ res,
      float* __restrict__ out, int Cout, int Npix) {
  __shared__ unsigned short sB[8 * 512];
  const int tid  = threadIdx.x;
  const int lane = tid & 31;
  const int wid  = tid >> 5;
  const int half = lane >> 4;
  const int nl   = lane & 15;
  const int p0   = blockIdx.x * 128;
  const int b    = blockIdx.y;

  for (int i = tid; i < 32 * 128; i += 256) {
    int pl = i & 127, c = i >> 7;
    float val = att[((size_t)b * 32 + c) * Npix + p0 + pl];
    int nt = pl >> 4, pnl = pl & 15;
    sB[nt * 512 + (pnl + ((c >= 16) ? 16 : 0)) * 16 + (c & 15)] = f2bf(val);
  }
  __syncthreads();

  const int nt = wid;
  v16bf bt = *(const v16bf*)(sB + nt * 512 + lane * 16);
  const int p = p0 + nt * 16 + nl;
  v8f z = {};
  for (int mt = 0; mt < (Cout >> 4); ++mt) {
    v16bf a = *(const v16bf*)(wsw + wbase + mt * 512 + lane * 16);
    v8f acc = wmma_bf16(a, bt, z);
    #pragma unroll
    for (int v = 0; v < 8; ++v) {
      int m = mt * 16 + v + half * 8;
      size_t idx = ((size_t)b * Cout + m) * Npix + p;
      out[idx] = acc[v] + bias[m] + res[idx];
    }
  }
}

// ---------------------------------------------------------------------------
extern "C" void kernel_launch(void* const* d_in, const int* in_sizes, int n_in,
                              void* d_out, int out_size, void* d_ws, size_t ws_size,
                              hipStream_t stream) {
  (void)in_sizes; (void)n_in; (void)out_size; (void)ws_size;
  const float* x    = (const float*)d_in[0];
  const float* y    = (const float*)d_in[1];
  const float* w_xc = (const float*)d_in[2];  const float* s_xc = (const float*)d_in[3];  const float* b_xc = (const float*)d_in[4];
  const float* w_qx = (const float*)d_in[5];  const float* s_qx = (const float*)d_in[6];  const float* b_qx = (const float*)d_in[7];
  const float* w_kx = (const float*)d_in[8];  const float* s_kx = (const float*)d_in[9];  const float* b_kx = (const float*)d_in[10];
  const float* w_vx = (const float*)d_in[11]; const float* s_vx = (const float*)d_in[12]; const float* b_vx = (const float*)d_in[13];
  const float* w_yc = (const float*)d_in[14]; const float* s_yc = (const float*)d_in[15]; const float* b_yc = (const float*)d_in[16];
  const float* w_qy = (const float*)d_in[17]; const float* s_qy = (const float*)d_in[18]; const float* b_qy = (const float*)d_in[19];
  const float* w_xa = (const float*)d_in[20]; const float* s_xa = (const float*)d_in[21]; const float* b_xa = (const float*)d_in[22];
  const float* w_ya = (const float*)d_in[23]; const float* s_ya = (const float*)d_in[24]; const float* b_ya = (const float*)d_in[25];

  // workspace layout (floats), then pre-swizzled bf16 weights
  float* wsf  = (float*)d_ws;
  float* Qx   = wsf;            float* Kx   = wsf + 262144;  float* Vxt = wsf + 524288;
  float* Qy   = wsf + 786432;   float* Ky   = wsf + 851968;  float* Vyt = wsf + 917504;
  float* attx = wsf + 983040;   float* atty = wsf + 1245184;
  unsigned short* wsw = (unsigned short*)(wsf + 1310720);

  // weight prep: fold BN scale, bf16, A-operand swizzle (element offsets)
  struct WDesc { const float* W; const float* s; int off, O, I; };
  const WDesc wl[8] = {
    {w_xc, s_xc,     0,  32, 256}, {w_yc, s_yc,  8192,  32, 512},
    {w_qx, s_qx, 24576,  32,  32}, {w_kx, s_kx, 25600,  32,  32},
    {w_vx, s_vx, 26624,  32,  32}, {w_qy, s_qy, 27648,  32,  32},
    {w_xa, s_xa, 28672, 256,  32}, {w_ya, s_ya, 36864, 512,  32}};
  for (int i = 0; i < 8; ++i) {
    int n = wl[i].O * wl[i].I;
    k_prep_w<<<(n + 255) / 256, 256, 0, stream>>>(wl[i].W, wl[i].s, wsw + wl[i].off,
                                                  wl[i].O, wl[i].I);
  }

  // projections (note reference quirk: y-branch K/V use w_kx / w_vx)
  k_proj<<<dim3(32, 2), 256, (8 * 8 * 512 + 8 * 512) * 2, stream>>>(
      x, wsw, 0, 24576, 25600, 26624, b_xc, b_qx, b_kx, b_vx,
      Qx, Kx, Vxt, 256, 4096);
  k_proj<<<dim3(8, 2), 256, (16 * 8 * 512 + 8 * 512) * 2, stream>>>(
      y, wsw, 8192, 27648, 25600, 26624, b_yc, b_qy, b_kx, b_vx,
      Qy, Ky, Vyt, 512, 1024);

  // flash cross-attention (x queries attend y; y queries attend x)
  k_attn<<<dim3(32, 8, 2), 256, 0, stream>>>(Qx, Ky, Vyt, attx, 4096, 1024);
  k_attn<<<dim3(8, 8, 2),  256, 0, stream>>>(Qy, Kx, Vxt, atty, 1024, 4096);

  // output convs + residual
  float* outx = (float*)d_out;
  float* outy = outx + 2 * 256 * 4096;
  k_out<<<dim3(32, 2), 256, 0, stream>>>(attx, wsw, 28672, b_xa, x, outx, 256, 4096);
  k_out<<<dim3(8, 2),  256, 0, stream>>>(atty, wsw, 36864, b_ya, y, outy, 512, 1024);
}
